// SSMDiTBlock_23175643529604
// MI455X (gfx1250) — matrix operand order repaired and checked by reading, not compiled, not run on hardware
//
#include <hip/hip_runtime.h>
#include <stdint.h>

// ---------------- problem constants ----------------
#define B_SZ   4
#define SEQ    4096
#define FEAT   1024
#define STATE  64
#define HID    4096
#define NB     256            // packed Bu/xs columns: {fwd_re, fwd_im, bwd_re, bwd_im} x 64
#define CHUNK  256
#define NCHUNK (SEQ / CHUNK)  // 16
#define MROWS  (B_SZ * SEQ)   // 16384
#define ADA6   (6 * FEAT)
#define MLP_CHUNK_ROWS 2048
#define N_MLP_CHUNKS   (MROWS / MLP_CHUNK_ROWS)

typedef __bf16 bf16x16 __attribute__((ext_vector_type(16)));
typedef __bf16 bf16x8  __attribute__((ext_vector_type(8)));
typedef float  f32x8   __attribute__((ext_vector_type(8)));

__device__ __forceinline__ uint16_t f2bf(float f) {
  union { float f; uint32_t u; } v; v.f = f;
  uint32_t u = v.u;
  uint32_t r = (u + 0x7FFFu + ((u >> 16) & 1u)) >> 16;   // RNE
  return (uint16_t)r;
}

// WMMA B-operand pack layout for W[K][N] (bf16), 16x16x32 tiles:
// lane = (c%16) + 16*((k/16)%2), element i = k%16, per-lane 16 contiguous bf16.
__device__ __forceinline__ size_t wpack_idx(int k, int c, int N) {
  int kt = k >> 5, nt = c >> 4;
  int lane = (c & 15) + (((k >> 4) & 1) << 4);
  int i = k & 15;
  return (((size_t)kt * (N >> 4) + nt) * 32 + lane) * 16 + i;
}

// ---------------- small precompute kernels ----------------

// ada[b][o] = silu(cond[b]) @ ada_W + ada_b
__global__ void k_ada(const float* __restrict__ cond, const float* __restrict__ W,
                      const float* __restrict__ bias, float* __restrict__ ada) {
  int o = blockIdx.x * blockDim.x + threadIdx.x;  // 0..6143
  int b = blockIdx.y;
  float acc = bias[o];
  for (int f = 0; f < FEAT; ++f) {
    float c = cond[b * FEAT + f];
    float s = c / (1.f + expf(-c));
    acc += s * W[(size_t)f * ADA6 + o];
  }
  ada[b * ADA6 + o] = acc;
}

// A_bar^k table: apows[dir][n][k] = exp(k*dt*A_diag), k = 0..CHUNK
__global__ void k_apows(const float* fLAr, const float* fAi, const float* fLdt,
                        const float* bLAr, const float* bAi, const float* bLdt,
                        float* __restrict__ apows) {
  int g = blockIdx.x * blockDim.x + threadIdx.x;
  if (g >= 2 * STATE * (CHUNK + 1)) return;
  int k = g % (CHUNK + 1);
  int n = (g / (CHUNK + 1)) % STATE;
  int dir = g / ((CHUNK + 1) * STATE);
  const float* LAr = dir ? bLAr : fLAr;
  const float* Ai  = dir ? bAi  : fAi;
  const float* Ldt = dir ? bLdt : fLdt;
  float dt = expf(Ldt[n]);
  float ar = -expf(LAr[n]);
  float ai = Ai[n];
  float e  = expf((float)k * dt * ar);
  float ph = (float)k * dt * ai;
  size_t idx = ((size_t)(dir * STATE + n) * (CHUNK + 1) + k) * 2;
  apows[idx + 0] = e * cosf(ph);
  apows[idx + 1] = e * sinf(ph);
}

// B_bar = (A_bar-1)/(A_diag+1e-8) * (B_re + i B_im), packed bf16 into W_B[K=FEAT][N=256]
__global__ void k_wb(const float* fLAr, const float* fAi, const float* fLdt,
                     const float* fBre, const float* fBim,
                     const float* bLAr, const float* bAi, const float* bLdt,
                     const float* bBre, const float* bBim,
                     uint16_t* __restrict__ WB) {
  int g = blockIdx.x * blockDim.x + threadIdx.x;  // 2*STATE*FEAT
  int f = g & (FEAT - 1);
  int n = (g >> 10) & (STATE - 1);
  int dir = g >> 16;
  const float* LAr = dir ? bLAr : fLAr;
  const float* Ai  = dir ? bAi  : fAi;
  const float* Ldt = dir ? bLdt : fLdt;
  const float* Bre = dir ? bBre : fBre;
  const float* Bim = dir ? bBim : fBim;
  float dt = expf(Ldt[n]);
  float ar = -expf(LAr[n]);
  float ai = Ai[n];
  float e = expf(dt * ar);
  float Abr = e * cosf(dt * ai), Abi = e * sinf(dt * ai);
  float dr = ar + 1e-8f, di = ai;
  float nr = Abr - 1.f, ni = Abi;
  float den = dr * dr + di * di;
  float sr = (nr * dr + ni * di) / den;
  float si = (ni * dr - nr * di) / den;
  float br = Bre[n * FEAT + f], bi = Bim[n * FEAT + f];
  float wr = sr * br - si * bi;
  float wi = sr * bi + si * br;
  WB[wpack_idx(f, dir * 128 + n, NB)]      = f2bf(wr);
  WB[wpack_idx(f, dir * 128 + 64 + n, NB)] = f2bf(wi);
}

// W_CO[c][g] = sgn * sum_f C[f,n] * out_W[dir*F+f][g]   (folds C-proj + out_W)
__global__ void k_wco(const float* fCre, const float* fCim,
                      const float* bCre, const float* bCim,
                      const float* __restrict__ outW, uint16_t* __restrict__ WCO) {
  int g = blockIdx.x * blockDim.x + threadIdx.x;  // 256*1024
  int col = g & (FEAT - 1);
  int c = g >> 10;
  int n = c & 63, part = (c >> 6) & 1, dir = c >> 7;
  const float* C = dir ? (part ? bCim : bCre) : (part ? fCim : fCre);
  float sgn = part ? -1.f : 1.f;
  float acc = 0.f;
  for (int f = 0; f < FEAT; ++f)
    acc += C[f * STATE + n] * outW[(size_t)(dir * FEAT + f) * FEAT + col];
  WCO[wpack_idx(c, col, FEAT)] = f2bf(sgn * acc);
}

// W_D[f][g] = D_f[f]*out_W[f][g] + D_b[f]*out_W[F+f][g]   (folds D-term + out_W)
__global__ void k_wd(const float* fD, const float* bD, const float* __restrict__ outW,
                     uint16_t* __restrict__ WD) {
  int g = blockIdx.x * blockDim.x + threadIdx.x;  // 1024*1024
  int col = g & (FEAT - 1);
  int f = g >> 10;
  float v = fD[f] * outW[(size_t)f * FEAT + col] + bD[f] * outW[(size_t)(FEAT + f) * FEAT + col];
  WD[wpack_idx(f, col, FEAT)] = f2bf(v);
}

// generic f32 row-major [K][N] -> packed bf16 WMMA-B layout
__global__ void k_packw(const float* __restrict__ W, uint16_t* __restrict__ Wp, int K, int N) {
  size_t g = (size_t)blockIdx.x * blockDim.x + threadIdx.x;
  if (g >= (size_t)K * N) return;
  int k = (int)(g / N), c = (int)(g % N);
  Wp[wpack_idx(k, c, N)] = f2bf(W[g]);
}

// ---------------- LayerNorm + AdaLN modulation -> bf16 ----------------
__global__ __launch_bounds__(256) void ln_mod(const float* __restrict__ src,
                                              uint16_t* __restrict__ dst,
                                              const float* __restrict__ ada,
                                              int shOff, int scOff) {
  int row = blockIdx.x;
  int b = row / SEQ;
  const float* p = src + (size_t)row * FEAT;
  int tid = threadIdx.x;
  float v[4], s = 0.f, ss = 0.f;
#pragma unroll
  for (int i = 0; i < 4; ++i) {
    float x = p[tid + i * 256];
    v[i] = x; s += x; ss += x * x;
  }
#pragma unroll
  for (int o = 16; o > 0; o >>= 1) { s += __shfl_xor(s, o, 32); ss += __shfl_xor(ss, o, 32); }
  __shared__ float sh[16];
  int wave = tid >> 5, lane = tid & 31;
  if (lane == 0) { sh[wave] = s; sh[wave + 8] = ss; }
  __syncthreads();
  float S = 0.f, SS = 0.f;
#pragma unroll
  for (int w = 0; w < 8; ++w) { S += sh[w]; SS += sh[w + 8]; }
  float mu = S * (1.f / FEAT);
  float var = SS * (1.f / FEAT) - mu * mu;
  float inv = rsqrtf(var + 1e-5f);
#pragma unroll
  for (int i = 0; i < 4; ++i) {
    int col = tid + i * 256;
    float sc = ada[b * ADA6 + scOff + col];
    float sf = ada[b * ADA6 + shOff + col];
    dst[(size_t)row * FEAT + col] = f2bf((v[i] - mu) * inv * (1.f + sc) + sf);
  }
}

// ---------------- WMMA GEMM (bf16 in, f32 acc), templated epilogue ----------------
// Wave tile 32Mx64N (2 A-frags x 4 B-frags -> 8 WMMA per k-tile, 24 B/WMMA).
// Block: 8 waves = 2 Mwaves x 4 Nwaves -> 64M x 256N.
// EPI 0: outF[m*N+c] = acc                                  (Bu)
// EPI 1: d_out = x + g1*(acc + bias)                        (S5 out-proj + residual)
// EPI 2: outBf = gelu_tanh(acc + bias)                      (MLP up)
// EPI 3: d_out += g2*(acc + bias)                           (MLP down + residual)
template <int EPI>
__global__ __launch_bounds__(256) void gemm_bf16(
    const uint16_t* __restrict__ A, int lda, const uint16_t* __restrict__ Wp, int K,
    const uint16_t* __restrict__ A2, int lda2, const uint16_t* __restrict__ Wp2, int K2,
    int N,
    const float* __restrict__ bias, const float* __restrict__ ada, int gateOff,
    const float* __restrict__ xres, float* __restrict__ outF, uint16_t* __restrict__ outBf,
    int rowBase) {
  const int tid = threadIdx.x, wave = tid >> 5, lane = tid & 31;
  const int half = lane >> 4, r = lane & 15;
  const int wm = wave & 1, wn = wave >> 1;
  const int mBase = blockIdx.y * 64 + wm * 32;
  const int nBase = blockIdx.x * 256 + wn * 64;
  const int Ntiles = N >> 4;

  f32x8 acc[2][4];
#pragma unroll
  for (int mi = 0; mi < 2; ++mi)
#pragma unroll
    for (int j = 0; j < 4; ++j)
#pragma unroll
      for (int v = 0; v < 8; ++v) acc[mi][j][v] = 0.f;

  for (int pass = 0; pass < 2; ++pass) {
    const uint16_t* Ac; const uint16_t* Wc; int Kc, ldc;
    if (pass == 0) { Ac = A; Wc = Wp; Kc = K; ldc = lda; }
    else { if (A2 == nullptr) break; Ac = A2; Wc = Wp2; Kc = K2; ldc = lda2; }
    const int nkt = Kc >> 5;
    for (int kt = 0; kt < nkt; ++kt) {
      // A fragments: rows (mBase+r) and (mBase+16+r); K = i + 16*half
      bf16x16 a[2];
#pragma unroll
      for (int mi = 0; mi < 2; ++mi) {
        const uint16_t* ap = Ac + (size_t)(mBase + mi * 16 + r) * ldc + kt * 32 + 16 * half;
        bf16x8 lo = *reinterpret_cast<const bf16x8*>(ap);
        bf16x8 hi = *reinterpret_cast<const bf16x8*>(ap + 8);
#pragma unroll
        for (int i = 0; i < 8; ++i) { a[mi][i] = lo[i]; a[mi][i + 8] = hi[i]; }
      }
      // unconditional prefetch of the B stream 4 k-tiles ahead
      // (speculative DEV-scope global_prefetch_b8; overshoot lands in live
      //  workspace buffers that follow every weight pack, so always mapped)
      {
        const uint16_t* pf =
            Wc + (((size_t)(kt + 4) * Ntiles + (nBase >> 4)) * 32 + lane) * 16;
        __builtin_prefetch((const void*)pf, 0, 1);
      }
#pragma unroll
      for (int j = 0; j < 4; ++j) {
        int nt = (nBase >> 4) + j;
        const uint16_t* bp = Wc + (((size_t)kt * Ntiles + nt) * 32 + lane) * 16;
        bf16x16 bfrag = *reinterpret_cast<const bf16x16*>(bp);
        acc[0][j] = __builtin_amdgcn_wmma_f32_16x16x32_bf16(
            false, a[0], false, bfrag, (short)0, acc[0][j], false, false);
        acc[1][j] = __builtin_amdgcn_wmma_f32_16x16x32_bf16(
            false, a[1], false, bfrag, (short)0, acc[1][j], false, false);
      }
    }
  }

#pragma unroll
  for (int mi = 0; mi < 2; ++mi) {
#pragma unroll
    for (int j = 0; j < 4; ++j) {
      int col = nBase + j * 16 + r;
      float bv = (EPI != 0) ? bias[col] : 0.f;
#pragma unroll
      for (int v = 0; v < 8; ++v) {
        int mI = mBase + mi * 16 + v + 8 * half;  // C/D: lanes0-15 M=v, lanes16-31 M=v+8
        float val = acc[mi][j][v];
        if (EPI == 0) {
          outF[(size_t)mI * N + col] = val;
        } else if (EPI == 1) {
          int grow = rowBase + mI;
          int b = grow / SEQ;
          float g = ada[b * ADA6 + gateOff + col];
          outF[(size_t)grow * N + col] = xres[(size_t)grow * N + col] + g * (val + bv);
        } else if (EPI == 2) {
          float x = val + bv;
          float t = tanhf(0.7978845608028654f * (x + 0.044715f * x * x * x));
          outBf[(size_t)mI * N + col] = f2bf(0.5f * x * (1.f + t));
        } else {
          int grow = rowBase + mI;
          int b = grow / SEQ;
          float g = ada[b * ADA6 + gateOff + col];
          outF[(size_t)grow * N + col] += g * (val + bv);
        }
      }
    }
  }
}

// ---------------- S5 chunked parallel scan (complex, fp32) ----------------
// phase 1: per-(dir,b,n,chunk) local scan (in place in Bu), record chunk-end state
__global__ void scan1(float* __restrict__ Bu, const float* __restrict__ apows,
                      float* __restrict__ cend) {
  int g = blockIdx.x * blockDim.x + threadIdx.x;  // 8192
  int n = g & 63, chunk = (g >> 6) & 15, b = (g >> 10) & 3, dir = g >> 12;
  const float* ap = apows + ((size_t)(dir * STATE + n) * (CHUNK + 1) + 1) * 2;
  float Ar = ap[0], Ai = ap[1];
  float xr = 0.f, xi = 0.f;
  for (int j = 0; j < CHUNK; ++j) {
    int t = chunk * CHUNK + j;
    int s = dir ? (SEQ - 1 - t) : t;
    size_t base = ((size_t)(b * SEQ + s)) * NB + dir * 128 + n;
    float br = Bu[base], bi = Bu[base + 64];
    float nr = Ar * xr - Ai * xi + br;
    float ni = Ar * xi + Ai * xr + bi;
    xr = nr; xi = ni;
    Bu[base] = xr; Bu[base + 64] = xi;
  }
  size_t ce = (((size_t)(dir * 4 + b) * NCHUNK + chunk) * STATE + n) * 2;
  cend[ce] = xr; cend[ce + 1] = xi;
}

// phase 2: per-(dir,b,n) exclusive scan of chunk carries
__global__ void scan2(const float* __restrict__ cend, float* __restrict__ cin,
                      const float* __restrict__ apows) {
  int g = blockIdx.x * blockDim.x + threadIdx.x;
  if (g >= 512) return;
  int n = g & 63, b = (g >> 6) & 3, dir = g >> 8;
  const float* ap = apows + ((size_t)(dir * STATE + n) * (CHUNK + 1) + CHUNK) * 2;
  float Pr = ap[0], Pi = ap[1];   // A_bar^CHUNK
  float pr = 0.f, pi = 0.f;
  for (int j = 0; j < NCHUNK; ++j) {
    size_t idx = (((size_t)(dir * 4 + b) * NCHUNK + j) * STATE + n) * 2;
    cin[idx] = pr; cin[idx + 1] = pi;
    float er = cend[idx], ei = cend[idx + 1];
    float nr = Pr * pr - Pi * pi + er;
    float ni = Pr * pi + Pi * pr + ei;
    pr = nr; pi = ni;
  }
}

// phase 3: xs = local + A^(pos+1) * carry ; emit bf16 for the C-projection GEMM
__global__ void scan3(const float* __restrict__ Bu, const float* __restrict__ cin,
                      const float* __restrict__ apows, uint16_t* __restrict__ xs) {
  int g = blockIdx.x * blockDim.x + threadIdx.x;  // 2^21
  int n = g & 63;
  int s = (g >> 6) & 4095;
  int b = (g >> 18) & 3;
  int dir = (g >> 20) & 1;
  int t = dir ? (SEQ - 1 - s) : s;
  int chunk = t >> 8, pos = t & 255;
  const float* ap = apows + ((size_t)(dir * STATE + n) * (CHUNK + 1) + (pos + 1)) * 2;
  float Ar = ap[0], Ai = ap[1];
  size_t ci = (((size_t)(dir * 4 + b) * NCHUNK + chunk) * STATE + n) * 2;
  float cr = cin[ci], cim = cin[ci + 1];
  size_t base = ((size_t)(b * SEQ + s)) * NB + dir * 128 + n;
  float lr = Bu[base], li = Bu[base + 64];
  xs[base]      = f2bf(lr + Ar * cr - Ai * cim);
  xs[base + 64] = f2bf(li + Ar * cim + Ai * cr);
}

// ---------------- host orchestration ----------------
extern "C" void kernel_launch(void* const* d_in, const int* in_sizes, int n_in,
                              void* d_out, int out_size, void* d_ws, size_t ws_size,
                              hipStream_t stream) {
  (void)in_sizes; (void)n_in; (void)out_size;
  const float* x     = (const float*)d_in[0];
  const float* cond  = (const float*)d_in[1];
  const float* adaW  = (const float*)d_in[2];
  const float* adab  = (const float*)d_in[3];
  const float* outW  = (const float*)d_in[4];
  const float* outb  = (const float*)d_in[5];
  const float* W1    = (const float*)d_in[6];
  const float* b1    = (const float*)d_in[7];
  const float* W2    = (const float*)d_in[8];
  const float* b2    = (const float*)d_in[9];
  const float* fLAr = (const float*)d_in[11]; const float* fAi  = (const float*)d_in[12];
  const float* fBre = (const float*)d_in[13]; const float* fBim = (const float*)d_in[14];
  const float* fCre = (const float*)d_in[15]; const float* fCim = (const float*)d_in[16];
  const float* fD   = (const float*)d_in[17]; const float* fLdt = (const float*)d_in[18];
  const float* bLAr = (const float*)d_in[19]; const float* bAi  = (const float*)d_in[20];
  const float* bBre = (const float*)d_in[21]; const float* bBim = (const float*)d_in[22];
  const float* bCre = (const float*)d_in[23]; const float* bCim = (const float*)d_in[24];
  const float* bD   = (const float*)d_in[25]; const float* bLdt = (const float*)d_in[26];
  float* out = (float*)d_out;
  uint8_t* ws = (uint8_t*)d_ws;

  // workspace layout (all 256B aligned)
  const size_t off_ada   = 0;                            // 4*6144 f32
  const size_t off_apows = off_ada   + 98304;            // 2*64*257*2 f32
  const size_t off_cend  = off_apows + 263168;           // 512*16*2 f32
  const size_t off_cin   = off_cend  + 65536;
  const size_t off_WB    = off_cin   + 65536;            // 1024x256 bf16
  const size_t off_WCO   = off_WB    + 524288;           // 256x1024 bf16
  const size_t off_WD    = off_WCO   + 524288;           // 1024x1024 bf16
  const size_t off_W1p   = off_WD    + 2097152;          // 1024x4096 bf16
  const size_t off_W2p   = off_W1p   + 8388608;          // 4096x1024 bf16
  const size_t off_h     = off_W2p   + 8388608;          // 16384x1024 bf16
  const size_t off_h2    = off_h     + 33554432;         // 16384x1024 bf16
  const size_t off_Bu    = off_h2    + 33554432;         // 16384x256 f32
  const size_t off_xs    = off_Bu    + 16777216;         // 16384x256 bf16
  const size_t off_mid   = off_xs    + 8388608;          // 2048x4096 bf16
  const size_t need      = off_mid   + 16777216;
  if (ws_size < need) return;  // scratch too small; do nothing (deterministic)

  float*    ada   = (float*)(ws + off_ada);
  float*    apows = (float*)(ws + off_apows);
  float*    cend  = (float*)(ws + off_cend);
  float*    cin   = (float*)(ws + off_cin);
  uint16_t* WB    = (uint16_t*)(ws + off_WB);
  uint16_t* WCO   = (uint16_t*)(ws + off_WCO);
  uint16_t* WD    = (uint16_t*)(ws + off_WD);
  uint16_t* W1p   = (uint16_t*)(ws + off_W1p);
  uint16_t* W2p   = (uint16_t*)(ws + off_W2p);
  uint16_t* h     = (uint16_t*)(ws + off_h);
  uint16_t* h2    = (uint16_t*)(ws + off_h2);
  float*    Bu    = (float*)(ws + off_Bu);
  uint16_t* xs    = (uint16_t*)(ws + off_xs);
  uint16_t* mid   = (uint16_t*)(ws + off_mid);

  // --- parameter precompute ---
  k_ada<<<dim3(ADA6 / 256, B_SZ), 256, 0, stream>>>(cond, adaW, adab, ada);
  k_apows<<<(2 * STATE * (CHUNK + 1) + 255) / 256, 256, 0, stream>>>(
      fLAr, fAi, fLdt, bLAr, bAi, bLdt, apows);
  k_wb<<<(2 * STATE * FEAT) / 256, 256, 0, stream>>>(
      fLAr, fAi, fLdt, fBre, fBim, bLAr, bAi, bLdt, bBre, bBim, WB);
  k_wco<<<(NB * FEAT) / 256, 256, 0, stream>>>(fCre, fCim, bCre, bCim, outW, WCO);
  k_wd<<<(FEAT * FEAT) / 256, 256, 0, stream>>>(fD, bD, outW, WD);
  k_packw<<<(FEAT * HID) / 256, 256, 0, stream>>>(W1, W1p, FEAT, HID);
  k_packw<<<(HID * FEAT) / 256, 256, 0, stream>>>(W2, W2p, HID, FEAT);

  // --- LN1 + modulation -> h (bf16) ---
  ln_mod<<<MROWS, 256, 0, stream>>>(x, h, ada, /*sh1*/ 0, /*sc1*/ FEAT);

  // --- Bu = h @ B_bar^T (both dirs, re/im packed into 256 cols) ---
  gemm_bf16<0><<<dim3(NB / 256, MROWS / 64), 256, 0, stream>>>(
      h, FEAT, WB, FEAT, nullptr, 0, nullptr, 0, NB,
      nullptr, nullptr, 0, nullptr, Bu, nullptr, 0);

  // --- bidirectional associative scan ---
  scan1<<<(2 * B_SZ * STATE * NCHUNK) / 256, 256, 0, stream>>>(Bu, apows, cend);
  scan2<<<2, 256, 0, stream>>>(cend, cin, apows);
  scan3<<<(2 * B_SZ * SEQ * STATE) / 256, 256, 0, stream>>>(Bu, cin, apows, xs);

  // --- y = xs@W_CO + h@W_D (+out_b), x1 = x + g1*y -> d_out ---
  gemm_bf16<1><<<dim3(FEAT / 256, MROWS / 64), 256, 0, stream>>>(
      xs, NB, WCO, NB, h, FEAT, WD, FEAT, FEAT,
      outb, ada, /*g1*/ 2 * FEAT, x, out, nullptr, 0);

  // --- LN2 + modulation -> h2 (bf16) ---
  ln_mod<<<MROWS, 256, 0, stream>>>(out, h2, ada, /*sh2*/ 3 * FEAT, /*sc2*/ 4 * FEAT);

  // --- MLP, 8 row-chunks: gelu(h2@W1+b1)@W2+b2 gated into d_out ---
  for (int ch = 0; ch < N_MLP_CHUNKS; ++ch) {
    const uint16_t* h2c = h2 + (size_t)ch * MLP_CHUNK_ROWS * FEAT;
    gemm_bf16<2><<<dim3(HID / 256, MLP_CHUNK_ROWS / 64), 256, 0, stream>>>(
        h2c, FEAT, W1p, FEAT, nullptr, 0, nullptr, 0, HID,
        b1, nullptr, 0, nullptr, nullptr, mid, 0);
    gemm_bf16<3><<<dim3(FEAT / 256, MLP_CHUNK_ROWS / 64), 256, 0, stream>>>(
        mid, HID, W2p, HID, nullptr, 0, nullptr, 0, FEAT,
        b2, ada, /*g2*/ 5 * FEAT, nullptr, out, nullptr, ch * MLP_CHUNK_ROWS);
  }
}
